// MMFP4Expert_27333171871987
// MI455X (gfx1250) — matrix operand order repaired and checked by reading, not compiled, not run on hardware
//
#include <hip/hip_runtime.h>
#include <stdint.h>

// ---------------------------------------------------------------------------
// MMFP4 SwiGLU MLP for gfx1250 (MI455X, wave32, WMMA)
//   out = (silu(x @ Wg^T) * (x @ Wu^T)) @ Wd^T
// FP4-E2M1 weight-only dequant -> f16 WMMA (f32 accum), double-buffered LDS.
// ---------------------------------------------------------------------------

typedef _Float16 v16h __attribute__((ext_vector_type(16)));
typedef _Float16 v8h  __attribute__((ext_vector_type(8)));
typedef _Float16 h2   __attribute__((ext_vector_type(2)));
typedef float    v8f  __attribute__((ext_vector_type(8)));

#define T_DIM 2048
#define H_DIM 4096
#define I_DIM 11008
#define KSTEP 64
#define KP2   72   // padded LDS K-stride in halves (144 B row: conflict-free)

// ---------------------------------------------------------------------------
// Packed FP4 dequant. E2M1 -> f16 identity: bits = (n&7)<<9 | (n&8)<<12 gives
// value * 2^-14 (e==0 maps onto the exact f16 subnormal; denorms are enabled).
// The 2^14 is folded into the group scale, applied with v_pk_mul_f16.
// One packed word (8 nibbles) -> 4 packed-f16 dwords -> one ds_store_b128.
// ---------------------------------------------------------------------------
__device__ __forceinline__ void dequant_store(uint32_t w, _Float16 hs,
                                              _Float16* dst) {
  h2 s2; s2[0] = hs; s2[1] = hs;
  uint32_t pr[4];
#pragma unroll
  for (int p = 0; p < 4; ++p) {
    const uint32_t b = w >> (8 * p);           // two nibbles in b[7:0]
    const uint32_t pair = ((b & 0x7u)  << 9)  | ((b & 0x8u)  << 12)
                        | ((b & 0x70u) << 21) | ((b & 0x80u) << 24);
    const h2 hv = __builtin_bit_cast(h2, pair) * s2;   // v_pk_mul_f16
    pr[p] = __builtin_bit_cast(uint32_t, hv);
  }
  uint4 q; q.x = pr[0]; q.y = pr[1]; q.z = pr[2]; q.w = pr[3];
  *(uint4*)dst = q;
}

// A fragment (16x32 f16): lane holds row (lane&15);
// halves 0-7 = K koff..+7, halves 8-15 = K koff+16..+23, koff=(lane>>4)*8
__device__ __forceinline__ v16h load_a_frag(const _Float16* tile, int lane) {
  const int row  = lane & 15;
  const int koff = (lane >> 4) << 3;
  const _Float16* p = tile + row * KP2 + koff;
  v8h lo = *(const v8h*)(p);
  v8h hi = *(const v8h*)(p + 16);
  return __builtin_shufflevector(lo, hi, 0,1,2,3,4,5,6,7,8,9,10,11,12,13,14,15);
}

// B fragment (32x16 f16): lane holds column (lane&15);
// halves 0-7 = K kb..+7, halves 8-15 = K kb+16..+23, kb=(lane>>4)*8. LDS tile stored [N][K].
__device__ __forceinline__ v16h load_b_frag(const _Float16* tile, int lane) {
  const int col = lane & 15;
  const int kb  = (lane >> 4) << 3;
  const _Float16* p = tile + col * KP2 + kb;
  v8h lo = *(const v8h*)(p);
  v8h hi = *(const v8h*)(p + 16);
  return __builtin_shufflevector(lo, hi, 0,1,2,3,4,5,6,7,8,9,10,11,12,13,14,15);
}

// ---------------------------------------------------------------------------
// Kernel 1: fused gate+up. Block tile 128(T) x 64(I), 8 waves, each wave a
// 32x32 sub-tile of BOTH matrices. K-phase = 64, LDS double-buffered.
// ---------------------------------------------------------------------------
__global__ __launch_bounds__(256)
void mmfp4_gateup_kernel(const float* __restrict__ x,
                         const uint32_t* __restrict__ gpw,
                         const float* __restrict__ gsc,
                         const uint32_t* __restrict__ upw,
                         const float* __restrict__ usc,
                         _Float16* __restrict__ hout) {
  __shared__ _Float16 As[2][128 * KP2];
  __shared__ _Float16 Bg[2][64 * KP2];
  __shared__ _Float16 Bu[2][64 * KP2];

  const int t    = threadIdx.x;
  const int lane = t & 31;
  const int wid  = t >> 5;
  const int mw   = (wid >> 1) * 32;
  const int nw   = (wid & 1) * 32;

  const int rowB = blockIdx.y * 128;  // T
  const int colB = blockIdx.x * 64;   // I

  // B-loader mapping: 64 rows x 8 words x 2 matrices -> 4 words/thread
  const int br   = t >> 2;            // 0..63 (N row)
  const int bw   = t & 3;             // word slot; this thread owns bw, bw+4
  const int bcol = colB + br;         // global I index

  v8f ag[2][2] = {};
  v8f au[2][2] = {};

  float4   aS[8];
  uint32_t gS[2], uS[2];
  _Float16 hsg = (_Float16)0.0f, hsu = (_Float16)0.0f;

  auto LOAD = [&](int k0) {
#pragma unroll
    for (int i = 0; i < 8; ++i) {
      const int c = t + i * 256, r = c >> 4, c4 = (c & 15) * 4;
      aS[i] = *(const float4*)(x + (size_t)(rowB + r) * H_DIM + k0 + c4);
    }
    const size_t wb = (size_t)bcol * (H_DIM / 8) + (k0 >> 3) + bw;
    gS[0] = gpw[wb]; gS[1] = gpw[wb + 4];
    uS[0] = upw[wb]; uS[1] = upw[wb + 4];
    const int kg = k0 >> 7;                       // 64 | 128 -> fixed group
    hsg = (_Float16)(gsc[(size_t)kg * I_DIM + bcol] * 16384.0f);
    hsu = (_Float16)(usc[(size_t)kg * I_DIM + bcol] * 16384.0f);
  };

  auto STORE = [&](int buf) {
#pragma unroll
    for (int i = 0; i < 8; ++i) {
      const int c = t + i * 256, r = c >> 4, c4 = (c & 15) * 4;
      h2* d = (h2*)&As[buf][r * KP2 + c4];
      h2 p0, p1;
      p0[0] = (_Float16)aS[i].x; p0[1] = (_Float16)aS[i].y;
      p1[0] = (_Float16)aS[i].z; p1[1] = (_Float16)aS[i].w;
      d[0] = p0; d[1] = p1;
    }
    dequant_store(gS[0], hsg, &Bg[buf][br * KP2 + bw * 8]);
    dequant_store(gS[1], hsg, &Bg[buf][br * KP2 + (bw + 4) * 8]);
    dequant_store(uS[0], hsu, &Bu[buf][br * KP2 + bw * 8]);
    dequant_store(uS[1], hsu, &Bu[buf][br * KP2 + (bw + 4) * 8]);
  };

  LOAD(0);
  STORE(0);
  __syncthreads();

  const int NK = H_DIM / KSTEP;       // 64
  for (int kk = 0; kk < NK; ++kk) {
    const int cur = kk & 1;
    if (kk + 1 < NK) LOAD((kk + 1) * KSTEP);   // overlap HBM with WMMA below

#pragma unroll
    for (int kc = 0; kc < 2; ++kc) {           // two 32-wide K chunks
      v16h af[2], bgf[2], buf_[2];
#pragma unroll
      for (int mt = 0; mt < 2; ++mt)
        af[mt] = load_a_frag(&As[cur][(mw + mt * 16) * KP2 + kc * 32], lane);
#pragma unroll
      for (int nt = 0; nt < 2; ++nt) {
        bgf[nt]  = load_b_frag(&Bg[cur][(nw + nt * 16) * KP2 + kc * 32], lane);
        buf_[nt] = load_b_frag(&Bu[cur][(nw + nt * 16) * KP2 + kc * 32], lane);
      }
#pragma unroll
      for (int mt = 0; mt < 2; ++mt) {
#pragma unroll
        for (int nt = 0; nt < 2; ++nt) {
          ag[mt][nt] = __builtin_amdgcn_wmma_f32_16x16x32_f16(
              false, af[mt], false, bgf[nt], (short)0, ag[mt][nt], false, false);
          au[mt][nt] = __builtin_amdgcn_wmma_f32_16x16x32_f16(
              false, af[mt], false, buf_[nt], (short)0, au[mt][nt], false, false);
        }
      }
    }

    if (kk + 1 < NK) STORE(1 - cur);           // fill next buffer
    __syncthreads();
  }

  // epilogue: h = silu(gate) * up -> f16 workspace [T, I]
  // C layout: VGPR v of lane L holds M = v + 8*(L>>4), N = L&15
#pragma unroll
  for (int mt = 0; mt < 2; ++mt) {
#pragma unroll
    for (int nt = 0; nt < 2; ++nt) {
#pragma unroll
      for (int v = 0; v < 8; ++v) {
        const float g = ag[mt][nt][v];
        const float u = au[mt][nt][v];
        const float hval = (g / (1.0f + __expf(-g))) * u;
        const int row = rowB + mw + mt * 16 + v + ((lane >> 4) << 3);
        const int col = colB + nw + nt * 16 + (lane & 15);
        hout[(size_t)row * I_DIM + col] = (_Float16)hval;
      }
    }
  }
}

// ---------------------------------------------------------------------------
// Kernel 2: down projection. Block tile 128(T) x 64(H), 8 waves x 32x32.
// A read from f16 workspace; K runs over I, phase = 64, double-buffered.
// ---------------------------------------------------------------------------
__global__ __launch_bounds__(256)
void mmfp4_down_kernel(const _Float16* __restrict__ h,
                       const uint32_t* __restrict__ dpw,
                       const float* __restrict__ dsc,
                       float* __restrict__ out) {
  __shared__ _Float16 As[2][128 * KP2];
  __shared__ _Float16 Bs[2][64 * KP2];

  const int t    = threadIdx.x;
  const int lane = t & 31;
  const int wid  = t >> 5;
  const int mw   = (wid >> 1) * 32;
  const int nw   = (wid & 1) * 32;

  const int rowB = blockIdx.y * 128;  // T
  const int colB = blockIdx.x * 64;   // H

  const int br   = t >> 2;
  const int bw   = t & 3;
  const int bcol = colB + br;         // global H index

  v8f acc[2][2] = {};

  uint4    aS[4];
  uint32_t dS[2];
  _Float16 hsd = (_Float16)0.0f;

  auto LOAD = [&](int k0) {
#pragma unroll
    for (int i = 0; i < 4; ++i) {
      const int c = t + i * 256, r = c >> 3, c8 = (c & 7) * 8;
      aS[i] = *(const uint4*)(h + (size_t)(rowB + r) * I_DIM + k0 + c8);
    }
    const size_t wb = (size_t)bcol * (I_DIM / 8) + (k0 >> 3) + bw;
    dS[0] = dpw[wb]; dS[1] = dpw[wb + 4];
    const int kg = k0 >> 7;
    hsd = (_Float16)(dsc[(size_t)kg * H_DIM + bcol] * 16384.0f);
  };

  auto STORE = [&](int buf) {
#pragma unroll
    for (int i = 0; i < 4; ++i) {
      const int c = t + i * 256, r = c >> 3, c8 = (c & 7) * 8;
      *(uint4*)&As[buf][r * KP2 + c8] = aS[i];
    }
    dequant_store(dS[0], hsd, &Bs[buf][br * KP2 + bw * 8]);
    dequant_store(dS[1], hsd, &Bs[buf][br * KP2 + (bw + 4) * 8]);
  };

  LOAD(0);
  STORE(0);
  __syncthreads();

  const int NK = I_DIM / KSTEP;       // 172
  for (int kk = 0; kk < NK; ++kk) {
    const int cur = kk & 1;
    if (kk + 1 < NK) LOAD((kk + 1) * KSTEP);

#pragma unroll
    for (int kc = 0; kc < 2; ++kc) {
      v16h af[2], bf[2];
#pragma unroll
      for (int mt = 0; mt < 2; ++mt)
        af[mt] = load_a_frag(&As[cur][(mw + mt * 16) * KP2 + kc * 32], lane);
#pragma unroll
      for (int nt = 0; nt < 2; ++nt)
        bf[nt] = load_b_frag(&Bs[cur][(nw + nt * 16) * KP2 + kc * 32], lane);
#pragma unroll
      for (int mt = 0; mt < 2; ++mt) {
#pragma unroll
        for (int nt = 0; nt < 2; ++nt) {
          acc[mt][nt] = __builtin_amdgcn_wmma_f32_16x16x32_f16(
              false, af[mt], false, bf[nt], (short)0, acc[mt][nt], false, false);
        }
      }
    }

    if (kk + 1 < NK) STORE(1 - cur);
    __syncthreads();
  }

  // epilogue: f32 output [T, H]
#pragma unroll
  for (int mt = 0; mt < 2; ++mt) {
#pragma unroll
    for (int nt = 0; nt < 2; ++nt) {
#pragma unroll
      for (int v = 0; v < 8; ++v) {
        const int row = rowB + mw + mt * 16 + v + ((lane >> 4) << 3);
        const int col = colB + nw + nt * 16 + (lane & 15);
        out[(size_t)row * H_DIM + col] = acc[mt][nt][v];
      }
    }
  }
}

// ---------------------------------------------------------------------------
extern "C" void kernel_launch(void* const* d_in, const int* in_sizes, int n_in,
                              void* d_out, int out_size, void* d_ws, size_t ws_size,
                              hipStream_t stream) {
  const float*    x    = (const float*)d_in[0];
  const uint32_t* gpw  = (const uint32_t*)d_in[1];
  const float*    gsc  = (const float*)d_in[2];
  const uint32_t* upw  = (const uint32_t*)d_in[3];
  const float*    usc  = (const float*)d_in[4];
  const uint32_t* dpw  = (const uint32_t*)d_in[5];
  const float*    dsc  = (const float*)d_in[6];
  float*          out  = (float*)d_out;
  _Float16*       hbuf = (_Float16*)d_ws;   // [T_DIM, I_DIM] f16 (~43 MB)

  dim3 blk(256);
  dim3 grid1(I_DIM / 64, T_DIM / 128);  // 172 x 16
  dim3 grid2(H_DIM / 64, T_DIM / 128);  // 64 x 16

  mmfp4_gateup_kernel<<<grid1, blk, 0, stream>>>(x, gpw, gsc, upw, usc, hbuf);
  mmfp4_down_kernel<<<grid2, blk, 0, stream>>>(hbuf, dpw, dsc, out);
}
